// MoE_33114197852627
// MI455X (gfx1250) — matrix operand order repaired and checked by reading, not compiled, not run on hardware
//
#include <hip/hip_runtime.h>
#include <hip/hip_bf16.h>

typedef __attribute__((ext_vector_type(16))) _Float16 v16h;
typedef __attribute__((ext_vector_type(8)))  _Float16 v8h;
typedef __attribute__((ext_vector_type(8)))  float    v8f;

namespace cfg {
constexpr int V  = 128;    // vocab
constexpr int E  = 512;    // embed
constexpr int H  = 1024;   // hidden
constexpr int NE = 10;     // experts
constexpr int B  = 32;     // batch
constexpr int S  = 1024;   // seq
constexpr int G  = 4 * H;  // 4096 gate width (i,f,g,o)
}

// ---------------------------------------------------------------------------
// WMMA tile loaders (v_wmma_f32_16x16x32_f16 operand layouts, ISA §7.12.2)
//
// A (16x32 f16, M rows): lane l<16 holds row M=l, halves {K0..7, K16..23};
// lane l+16 holds row M=l, halves {K8..15, K24..31}.  -> two b128 loads.
// ---------------------------------------------------------------------------
__device__ __forceinline__ v16h load_a16(const _Float16* __restrict__ arow,
                                         int kbase, int ksel) {
  v8h lo = *(const v8h*)(arow + kbase + ksel);
  v8h hi = *(const v8h*)(arow + kbase + 16 + ksel);
  v16h r;
#pragma unroll
  for (int i = 0; i < 8; ++i) { r[i] = lo[i]; r[i + 8] = hi[i]; }
  return r;
}

// B (32x16 f16): lane l holds column N=l%16, K in the same order as A
// ({K0..7,K16..23} for l<16, {K8..15,K24..31} for l>=16); strided gather.
__device__ __forceinline__ v16h load_b16(const _Float16* __restrict__ brow,
                                         int ld, int ksel) {
  v16h r;
#pragma unroll
  for (int i = 0; i < 8; ++i) {
    r[i] = brow[(size_t)(ksel + i) * ld];
    r[i + 8] = brow[(size_t)(16 + ksel + i) * ld];
  }
  return r;
}

__device__ __forceinline__ float sigm(float x) { return 1.0f / (1.0f + expf(-x)); }

// ---------------------------------------------------------------------------
// Embedding gather + f32->f16 (vectorized: one v8h store / thread):
//   xe[b,s,e] = f16(emb[x[b,s], e])
// ---------------------------------------------------------------------------
__global__ void k_embed(const int* __restrict__ x, const float* __restrict__ emb,
                        _Float16* __restrict__ xe) {
  using namespace cfg;
  size_t idx  = (size_t)blockIdx.x * blockDim.x + threadIdx.x;  // B*S*E/8
  size_t base = idx * 8;
  int tokIdx = (int)(base >> 9);          // / E (E==512)
  int e      = (int)(base & (E - 1));
  int tok    = x[tokIdx];
  const float4* src = (const float4*)(emb + (size_t)tok * E + e);
  float4 f0 = src[0], f1 = src[1];
  v8h o;
  o[0] = (_Float16)f0.x; o[1] = (_Float16)f0.y;
  o[2] = (_Float16)f0.z; o[3] = (_Float16)f0.w;
  o[4] = (_Float16)f1.x; o[5] = (_Float16)f1.y;
  o[6] = (_Float16)f1.z; o[7] = (_Float16)f1.w;
  *(v8h*)(xe + base) = o;
}

// Convert f32 weight [N][K] (row-major, K contiguous) -> f16 K-major [K][N].
__global__ void k_cvtT(const float* __restrict__ src, _Float16* __restrict__ dst,
                       int N, int K) {
  size_t idx = (size_t)blockIdx.x * blockDim.x + threadIdx.x;
  if (idx >= (size_t)N * K) return;
  int n = (int)(idx / K);
  int k = (int)(idx - (size_t)n * K);
  dst[(size_t)k * N + n] = (_Float16)src[idx];
}

// Zero initial hidden (f16) and cell (f32) state each call (deterministic).
__global__ void k_init(_Float16* __restrict__ hz, float* __restrict__ c) {
  int idx = blockIdx.x * blockDim.x + threadIdx.x;  // B*H
  hz[idx] = (_Float16)0.0f;
  c[idx]  = 0.0f;
}

// ---------------------------------------------------------------------------
// One LSTM step's gate pre-activations (no biases):
//   gbuf[B,4H] = [xe_t | h_prev] @ [W_ih ; W_hh]^T      (M=32,K=1536,N=4096)
// One wave per N-tile, covering BOTH M-tiles (B=32): each B-fragment is
// loaded once and feeds two independent WMMAs (pipelined D->C chains).
// 96 v_wmma per wave; grid = 256 waves.
// ---------------------------------------------------------------------------
__global__ void k_step_gemm(const _Float16* __restrict__ xe_t, long long xeld,
                            const _Float16* __restrict__ hprev, long long hld,
                            const _Float16* __restrict__ WihT,   // [E][G]
                            const _Float16* __restrict__ WhhT,   // [H][G]
                            float* __restrict__ gbuf) {
  using namespace cfg;
  int wid   = (blockIdx.x * blockDim.x + threadIdx.x) >> 5;  // 0..255 = ntile
  int lane  = threadIdx.x & 31;
  int l     = lane & 15;
  int ksel  = (lane >> 4) << 3;
  int nbase = wid << 4;

  const _Float16* a10 = xe_t  + (size_t)l * xeld;         // M rows 0..15
  const _Float16* a11 = xe_t  + (size_t)(16 + l) * xeld;  // M rows 16..31
  const _Float16* a20 = hprev + (size_t)l * hld;
  const _Float16* a21 = hprev + (size_t)(16 + l) * hld;
  const _Float16* b1  = WihT + nbase + l;   // B column N = l
  const _Float16* b2  = WhhT + nbase + l;

  v8f acc0 = {}, acc1 = {};
#pragma unroll 4
  for (int k = 0; k < E; k += 32) {
    v16h bm = load_b16(b1 + (size_t)k * G, G, ksel);
    acc0 = __builtin_amdgcn_wmma_f32_16x16x32_f16(false, load_a16(a10, k, ksel),
                                                  false, bm, (short)0, acc0,
                                                  false, false);
    acc1 = __builtin_amdgcn_wmma_f32_16x16x32_f16(false, load_a16(a11, k, ksel),
                                                  false, bm, (short)0, acc1,
                                                  false, false);
  }
#pragma unroll 4
  for (int k = 0; k < H; k += 32) {
    v16h bm = load_b16(b2 + (size_t)k * G, G, ksel);
    acc0 = __builtin_amdgcn_wmma_f32_16x16x32_f16(false, load_a16(a20, k, ksel),
                                                  false, bm, (short)0, acc0,
                                                  false, false);
    acc1 = __builtin_amdgcn_wmma_f32_16x16x32_f16(false, load_a16(a21, k, ksel),
                                                  false, bm, (short)0, acc1,
                                                  false, false);
  }

  // D layout: lane l<16 -> rows M=0..7 (VGPR r), lane l+16 -> rows M=8..15.
  int col = nbase + l;
#pragma unroll
  for (int r = 0; r < 8; ++r)
    gbuf[(size_t)(ksel + r) * G + col] = acc0[r];
#pragma unroll
  for (int r = 0; r < 8; ++r)
    gbuf[(size_t)(16 + ksel + r) * G + col] = acc1[r];
}

// Gate nonlinearities + state update; writes h_t (f16) into h_all[:, t, :].
__global__ void k_step_update(const float* __restrict__ gbuf,
                              const float* __restrict__ b_ih,
                              const float* __restrict__ b_hh,
                              float* __restrict__ cbuf,
                              _Float16* __restrict__ hout /* h_all + t*H */) {
  using namespace cfg;
  int idx = blockIdx.x * blockDim.x + threadIdx.x;  // B*H
  int b = idx >> 10;
  int j = idx & (H - 1);
  const float* g = gbuf + (size_t)b * G;
  float gi = g[j]           + b_ih[j]           + b_hh[j];
  float gf = g[H + j]       + b_ih[H + j]       + b_hh[H + j];
  float gg = g[2 * H + j]   + b_ih[2 * H + j]   + b_hh[2 * H + j];
  float go = g[3 * H + j]   + b_ih[3 * H + j]   + b_hh[3 * H + j];
  float c = sigm(gf) * cbuf[idx] + sigm(gi) * tanhf(gg);
  cbuf[idx] = c;
  hout[(size_t)b * ((size_t)S * H) + j] = (_Float16)(sigm(go) * tanhf(c));
}

// ---------------------------------------------------------------------------
// Gating:  gw = softmax(h_last @ Wg^T + bg)   (B=32 x NE=10, K=H)
// ---------------------------------------------------------------------------
__global__ void k_gating(const _Float16* __restrict__ h_last, long long hld,
                         const float* __restrict__ Wg, const float* __restrict__ bg,
                         float* __restrict__ gw_ws, float* __restrict__ gw_out) {
  using namespace cfg;
  __shared__ float logits[B * NE];
  int t = threadIdx.x;  // blockDim = 320
  if (t < B * NE) {
    int b = t / NE, e = t - b * NE;
    const _Float16* hp = h_last + (size_t)b * hld;
    const float* wp = Wg + (size_t)e * H;
    float s = bg[e];
    for (int k = 0; k < H; ++k) s += (float)hp[k] * wp[k];
    logits[t] = s;
  }
  __syncthreads();
  if (t < B) {
    float mx = logits[t * NE];
    for (int e = 1; e < NE; ++e) mx = fmaxf(mx, logits[t * NE + e]);
    float ex[NE];
    float den = 0.0f;
    for (int e = 0; e < NE; ++e) { ex[e] = expf(logits[t * NE + e] - mx); den += ex[e]; }
    for (int e = 0; e < NE; ++e) {
      float w = ex[e] / den;
      gw_ws[t * NE + e]  = w;
      gw_out[t * NE + e] = w;
    }
  }
}

// ---------------------------------------------------------------------------
// Expert GEMM: eo[e, m, v] = h_all[m,:] . We[e,v,:] + be[e,v]
// M = B*S = 32768, N = V = 128, K = H = 1024 per expert.
// One wave computes a full 16x128 strip (all 8 V-tiles of one M-tile):
// each A-fragment load feeds 8 WMMAs; the lane's 8 B-fragments are one
// contiguous 256B row of WeT. 256 v_wmma per wave; 20,480 waves total.
// ---------------------------------------------------------------------------
__global__ void k_expert_gemm(const _Float16* __restrict__ h_all,   // [B*S][H]
                              const _Float16* __restrict__ WeT,     // [NE][H][V]
                              const float* __restrict__ be,         // [NE][V]
                              float* __restrict__ eo) {             // [NE][B*S][V]
  using namespace cfg;
  int wid   = (blockIdx.x * blockDim.x + threadIdx.x) >> 5;
  int lane  = threadIdx.x & 31;
  int mtile = wid & 2047;           // B*S/16 = 2048
  int e     = wid >> 11;            // expert
  int l     = lane & 15;
  int ksel  = (lane >> 4) << 3;

  const _Float16* arow  = h_all + (size_t)(mtile * 16 + l) * H;
  const _Float16* bbase = WeT + (size_t)e * H * V + l;

  v8f acc[8] = {};
#pragma unroll 2
  for (int k = 0; k < H; k += 32) {
    v16h a = load_a16(arow, k, ksel);
    const _Float16* bk = bbase + (size_t)k * V;
#pragma unroll
    for (int vt = 0; vt < 8; ++vt) {
      v16h bm = load_b16(bk + vt * 16, V, ksel);
      acc[vt] = __builtin_amdgcn_wmma_f32_16x16x32_f16(false, a, false, bm,
                                                       (short)0, acc[vt],
                                                       false, false);
    }
  }

  int rbase = mtile * 16 + ksel;
  float* oute = eo + (size_t)e * B * S * V;
#pragma unroll
  for (int vt = 0; vt < 8; ++vt) {
    int col = vt * 16 + l;
    float bias = be[e * V + col];
#pragma unroll
    for (int r = 0; r < 8; ++r)
      oute[(size_t)(rbase + r) * V + col] = acc[vt][r] + bias;
  }
}

// combined[b,s,v] = sum_e gw[b,e] * eo[e,b,s,v]   (bandwidth-bound stream)
__global__ void k_combine(const float* __restrict__ eo, const float* __restrict__ gw,
                          float* __restrict__ outc) {
  using namespace cfg;
  size_t idx = (size_t)blockIdx.x * blockDim.x + threadIdx.x;  // B*S*V
  int b = (int)(idx / ((size_t)S * V));
  float s = 0.0f;
#pragma unroll
  for (int e = 0; e < NE; ++e)
    s += gw[b * NE + e] * eo[(size_t)e * B * S * V + idx];
  outc[idx] = s;
}

// ---------------------------------------------------------------------------
extern "C" void kernel_launch(void* const* d_in, const int* in_sizes, int n_in,
                              void* d_out, int out_size, void* d_ws, size_t ws_size,
                              hipStream_t stream) {
  using namespace cfg;
  (void)in_sizes; (void)n_in; (void)out_size; (void)ws_size;

  const int*   x    = (const int*)  d_in[0];
  const float* emb  = (const float*)d_in[1];
  const float* W_ih = (const float*)d_in[2];
  const float* W_hh = (const float*)d_in[3];
  const float* b_ih = (const float*)d_in[4];
  const float* b_hh = (const float*)d_in[5];
  const float* We   = (const float*)d_in[6];
  const float* be   = (const float*)d_in[7];
  const float* Wg   = (const float*)d_in[8];
  const float* bg   = (const float*)d_in[9];

  float* out_combined = (float*)d_out;                              // [B,S,V]
  float* out_experts  = out_combined + (size_t)B * S * V;           // [NE,B,S,V]
  float* out_gate     = out_experts + (size_t)NE * B * S * V;       // [B,NE]

  // Workspace carve (256B aligned): ~117 MB total.
  char* wsp = (char*)d_ws;
  auto carve = [&](size_t bytes) {
    char* p = wsp;
    wsp += (bytes + 255) & ~(size_t)255;
    return p;
  };
  _Float16* xe    = (_Float16*)carve((size_t)B * S * E * 2);   // 32 MB
  _Float16* WihT  = (_Float16*)carve((size_t)E * G * 2);       //  4 MB
  _Float16* WhhT  = (_Float16*)carve((size_t)H * G * 2);       //  8 MB
  _Float16* WeT   = (_Float16*)carve((size_t)NE * H * V * 2);  //2.5 MB
  _Float16* h_all = (_Float16*)carve((size_t)B * S * H * 2);   // 64 MB
  _Float16* hzero = (_Float16*)carve((size_t)B * H * 2);
  float*    cbuf  = (float*)   carve((size_t)B * H * 4);
  float*    gbuf  = (float*)   carve((size_t)B * G * 4);
  float*    gw    = (float*)   carve((size_t)B * NE * 4);

  // Preprocess: gather+cast embeddings; cast+transpose weights to K-major f16.
  k_embed<<<(B * S * E / 8) / 256, 256, 0, stream>>>(x, emb, xe);
  k_cvtT <<<(G * E + 255) / 256, 256, 0, stream>>>(W_ih, WihT, G, E);
  k_cvtT <<<(G * H + 255) / 256, 256, 0, stream>>>(W_hh, WhhT, G, H);
  for (int e = 0; e < NE; ++e)
    k_cvtT<<<(V * H + 255) / 256, 256, 0, stream>>>(
        We + (size_t)e * V * H, WeT + (size_t)e * H * V, V, H);
  k_init<<<(B * H) / 256, 256, 0, stream>>>(hzero, cbuf);

  // Sequential LSTM scan: fused [xe_t | h_prev] GEMM + gate update per step.
  for (int t = 0; t < S; ++t) {
    const _Float16* hprev = (t == 0) ? hzero : (h_all + (size_t)(t - 1) * H);
    long long hld = (t == 0) ? (long long)H : (long long)S * H;
    k_step_gemm<<<(G / 16) / 8, 256, 0, stream>>>(
        xe + (size_t)t * E, (long long)S * E, hprev, hld, WihT, WhhT, gbuf);
    k_step_update<<<(B * H) / 256, 256, 0, stream>>>(
        gbuf, b_ih, b_hh, cbuf, h_all + (size_t)t * H);
  }

  // Gating softmax from final hidden state.
  k_gating<<<1, 320, 0, stream>>>(h_all + (size_t)(S - 1) * H, (long long)S * H,
                                  Wg, bg, gw, out_gate);

  // All experts as one batched WMMA GEMM straight into d_out.
  int expert_waves = NE * ((B * S) / 16);  // 20,480 (one 16x128 strip each)
  k_expert_gemm<<<expert_waves / 8, 256, 0, stream>>>(h_all, WeT, be, out_experts);

  // Weighted combination of expert outputs.
  k_combine<<<(B * S * V) / 256, 256, 0, stream>>>(out_experts, gw, out_combined);
}
